// AnchorHead_8959301780158
// MI455X (gfx1250) — matrix operand order repaired and checked by reading, not compiled, not run on hardware
//
#include <hip/hip_runtime.h>
#include <hip/hip_bf16.h>

// ---------------------------------------------------------------------------
// AnchorHead for MI455X (gfx1250, wave32).
//   agent_anchor = LN(x*score)->ReLU @ aa_W + aa_b      (32768x1024x1024 GEMM)
//   scene_anchor = LN(maskmean(x*score))->ReLU @ sp_W + sp_b (256x1024x1024)
// GEMMs: v_wmma_f32_16x16x32_f16, LDS double-buffered, staged with
// global_load_async_to_lds_b128 (ASYNCcnt), B fragments software-pipelined.
// ---------------------------------------------------------------------------

typedef _Float16 v4h  __attribute__((ext_vector_type(4)));
typedef _Float16 v8h  __attribute__((ext_vector_type(8)));
typedef _Float16 v16h __attribute__((ext_vector_type(16)));
typedef float    v8f  __attribute__((ext_vector_type(8)));

#define D 1024
#define AGENTS 128
#define BATCH 256

// ---- CDNA5 async global->LDS copy (16B per lane), tracked on ASYNCcnt -----
__device__ __forceinline__ void async_b128(unsigned lds_addr, const void* gaddr) {
  asm volatile("global_load_async_to_lds_b128 %0, %1, off"
               :: "v"(lds_addr), "v"(gaddr) : "memory");
}
__device__ __forceinline__ void wait_async0() {
  asm volatile("s_wait_asynccnt 0" ::: "memory");
}

// ---------------------------------------------------------------------------
// Kernel 1: W[k][n] (f32 row-major) -> Wt[n][k] (f16 N-major), 32x32 LDS tiles
// ---------------------------------------------------------------------------
__global__ __launch_bounds__(256) void transpose_cvt_f16(
    const float* __restrict__ W, _Float16* __restrict__ Wt) {
  __shared__ float tile[32][33];
  const int k0 = blockIdx.y * 32, n0 = blockIdx.x * 32;
  const int tx = threadIdx.x, ty = threadIdx.y;       // block (32,8)
  for (int i = ty; i < 32; i += 8)
    tile[i][tx] = W[(size_t)(k0 + i) * D + n0 + tx];
  __syncthreads();
  for (int i = ty; i < 32; i += 8)
    Wt[(size_t)(n0 + i) * D + k0 + tx] = (_Float16)tile[tx][i];
}

// ---------------------------------------------------------------------------
// Kernel 2/4: per-row  h = relu(LN(x*scale)*gamma+beta) -> f16
// ---------------------------------------------------------------------------
__global__ __launch_bounds__(256) void scale_ln_relu_f16(
    const float* __restrict__ X, const float* __restrict__ scale,
    const float* __restrict__ gamma, const float* __restrict__ beta,
    _Float16* __restrict__ H) {
  const int row = blockIdx.x;
  const int t = threadIdx.x;
  const float s = scale ? scale[row] : 1.0f;

  float4 v = ((const float4*)X)[(size_t)row * 256 + t];
  v.x *= s; v.y *= s; v.z *= s; v.w *= s;

  __shared__ float2 red[256];
  red[t] = make_float2(v.x + v.y + v.z + v.w,
                       v.x * v.x + v.y * v.y + v.z * v.z + v.w * v.w);
  __syncthreads();
  for (int off = 128; off > 0; off >>= 1) {
    if (t < off) {
      red[t].x += red[t + off].x;
      red[t].y += red[t + off].y;
    }
    __syncthreads();
  }
  const float mean = red[0].x * (1.0f / (float)D);
  const float var  = red[0].y * (1.0f / (float)D) - mean * mean;
  const float rstd = rsqrtf(var + 1e-5f);

  const int d = 4 * t;
  v4h o;
  o.x = (_Float16)fmaxf(0.0f, (v.x - mean) * rstd * gamma[d + 0] + beta[d + 0]);
  o.y = (_Float16)fmaxf(0.0f, (v.y - mean) * rstd * gamma[d + 1] + beta[d + 1]);
  o.z = (_Float16)fmaxf(0.0f, (v.z - mean) * rstd * gamma[d + 2] + beta[d + 2]);
  o.w = (_Float16)fmaxf(0.0f, (v.w - mean) * rstd * gamma[d + 3] + beta[d + 3]);
  *(v4h*)&H[(size_t)row * D + d] = o;
}

// ---------------------------------------------------------------------------
// Kernel 3: masked mean pool over agents: pooled[b][d] = sum_a m*s*x / sum_a m
// ---------------------------------------------------------------------------
__global__ __launch_bounds__(256) void pool_kernel(
    const float* __restrict__ X, const unsigned char* __restrict__ mask,
    const float* __restrict__ score, float* __restrict__ pooled) {
  const int b = blockIdx.x, t = threadIdx.x;
  const float* xb = X + (size_t)b * AGENTS * D;
  float4 acc = make_float4(0.f, 0.f, 0.f, 0.f);
  float cnt = 0.f;
  for (int a = 0; a < AGENTS; ++a) {
    const float m = mask[b * AGENTS + a] ? 1.0f : 0.0f;
    cnt += m;
    const float w = m * score[b * AGENTS + a];
    float4 v = ((const float4*)(xb + (size_t)a * D))[t];
    acc.x += w * v.x; acc.y += w * v.y; acc.z += w * v.z; acc.w += w * v.w;
  }
  const float inv = (cnt > 0.f) ? (1.0f / cnt) : 0.0f;
  ((float4*)(pooled + (size_t)b * D))[t] =
      make_float4(acc.x * inv, acc.y * inv, acc.z * inv, acc.w * inv);
}

// ---------------------------------------------------------------------------
// Kernel 5: GEMM  out[M][1024] = H[M][1024](f16) @ W + bias  (f32 out)
// Block tile 64(M) x 256(N), 256 thr = 8 waves; wave grid 4(M) x 2(N):
// each wave -> 16x128 = 8 wmma tiles. K staged 32-wide through
// double-buffered LDS via async global->LDS copies; B fragments prefetched
// one tile ahead so ds latency hides under the matrix pipe.
// Fragment layouts per CDNA5 ISA 7.12.2 (16-bit A 16x32 / B 32x16).
// ---------------------------------------------------------------------------
#define LSTR 40  // padded LDS row stride in f16 (80B: 16B-aligned, bank-spread)

__global__ __launch_bounds__(256) void gemm_wmma_f16(
    const _Float16* __restrict__ H, const _Float16* __restrict__ Wt,
    const float* __restrict__ bias, float* __restrict__ out) {
  __shared__ __align__(16) _Float16 As[2][64 * LSTR];
  __shared__ __align__(16) _Float16 Bs[2][256 * LSTR];

  const int t = threadIdx.x;
  const int wave = t >> 5, lane = t & 31;
  const int half = lane >> 4, ln = lane & 15;
  const int wm = wave & 3, wn = wave >> 2;
  const int blockM = blockIdx.x * 64;
  const int n0 = blockIdx.y * 256;

  // --- per-thread async staging addresses ---------------------------------
  const int arow = t >> 2, achk = (t & 3) * 8;       // A: 1 x 16B per thread
  const _Float16* gA = &H[(size_t)(blockM + arow) * D + achk];
  const unsigned ldsA[2] = {(unsigned)(uintptr_t)&As[0][arow * LSTR + achk],
                            (unsigned)(uintptr_t)&As[1][arow * LSTR + achk]};

  const _Float16* gB[4];                             // B: 4 x 16B per thread
  unsigned ldsB[2][4];
#pragma unroll
  for (int j = 0; j < 4; ++j) {
    const int id = t + j * 256;
    const int brow = id >> 2, bchk = (id & 3) * 8;
    gB[j] = &Wt[(size_t)(n0 + brow) * D + bchk];
    ldsB[0][j] = (unsigned)(uintptr_t)&Bs[0][brow * LSTR + bchk];
    ldsB[1][j] = (unsigned)(uintptr_t)&Bs[1][brow * LSTR + bchk];
  }

  v8f acc[8] = {};

  // prologue: fill buffer 0
  async_b128(ldsA[0], gA);
#pragma unroll
  for (int j = 0; j < 4; ++j) async_b128(ldsB[0][j], gB[j]);
  wait_async0();
  __syncthreads();

#pragma unroll 2   // makes double-buffer index a compile-time constant
  for (int k0 = 0; k0 < D; k0 += 32) {
    const int cur = (k0 >> 5) & 1;
    const bool more = (k0 + 32) < D;
    if (more) {                          // async copies fly under the WMMAs
      async_b128(ldsA[cur ^ 1], gA + k0 + 32);
#pragma unroll
      for (int j = 0; j < 4; ++j)
        async_b128(ldsB[cur ^ 1][j], gB[j] + k0 + 32);
    }

    const _Float16* Ab = As[cur];
    const _Float16* Bb = Bs[cur];

    // A fragment (16x32): lanes 0-15 = rows, K lo/hi split across lane halves
    const _Float16* ap = &Ab[(wm * 16 + ln) * LSTR + half * 8];
    v8h alo = *(const v8h*)ap;          // K = half*8 .. +7   -> VGPR 0-3
    v8h ahi = *(const v8h*)(ap + 16);   // K = half*8+16..+23 -> VGPR 4-7
    v16h afrag = __builtin_shufflevector(alo, ahi, 0, 1, 2, 3, 4, 5, 6, 7, 8,
                                         9, 10, 11, 12, 13, 14, 15);

    // B fragments (32x16): lane = column, K = half*8..+7 and half*8+16..+23.
    // Prefetch tile t+1 before WMMA(t) so ds latency overlaps the matrix pipe.
    const _Float16* bbase = &Bb[(wn * 128 + ln) * LSTR + half * 8];
    v8h blo = *(const v8h*)bbase;
    v8h bhi = *(const v8h*)(bbase + 16);
#pragma unroll
    for (int tile = 0; tile < 8; ++tile) {
      v16h bfrag = __builtin_shufflevector(blo, bhi, 0, 1, 2, 3, 4, 5, 6, 7,
                                           8, 9, 10, 11, 12, 13, 14, 15);
      if (tile < 7) {
        const _Float16* bp = bbase + (tile + 1) * 16 * LSTR;
        blo = *(const v8h*)bp;
        bhi = *(const v8h*)(bp + 16);
      }
      acc[tile] = __builtin_amdgcn_wmma_f32_16x16x32_f16(
          false, afrag, false, bfrag, (short)0, acc[tile], false, false);
    }

    if (more) wait_async0();   // next slab resident in LDS
    __syncthreads();           // all waves done reading cur / writes visible
  }

  // --- epilogue: D tile layout -> lane n = N, VGPR r -> M = r (+8 hi half) --
  const int gm = blockM + wm * 16 + half * 8;
#pragma unroll
  for (int tile = 0; tile < 8; ++tile) {
    const int gn = n0 + wn * 128 + tile * 16 + ln;
    const float bv = bias[gn];
#pragma unroll
    for (int r = 0; r < 8; ++r)
      out[(size_t)(gm + r) * D + gn] = acc[tile][r] + bv;
  }
}

// ---------------------------------------------------------------------------
// Launch
// ---------------------------------------------------------------------------
extern "C" void kernel_launch(void* const* d_in, const int* in_sizes, int n_in,
                              void* d_out, int out_size, void* d_ws,
                              size_t ws_size, hipStream_t stream) {
  const float* encoder_feat = (const float*)d_in[0];
  const unsigned char* agent_mask = (const unsigned char*)d_in[1];  // jnp bool
  const float* agent_score = (const float*)d_in[2];
  const float* aa_gamma = (const float*)d_in[3];
  const float* aa_beta  = (const float*)d_in[4];
  const float* aa_W     = (const float*)d_in[5];
  const float* aa_b     = (const float*)d_in[6];
  const float* sp_gamma = (const float*)d_in[7];
  const float* sp_beta  = (const float*)d_in[8];
  const float* sp_W     = (const float*)d_in[9];
  const float* sp_b     = (const float*)d_in[10];

  float* agent_anchor = (float*)d_out;                         // [256,128,1024]
  float* scene_anchor = (float*)d_out + (size_t)BATCH * AGENTS * D;  // [256,1024]

  // workspace layout (~73 MB)
  char* ws = (char*)d_ws;
  _Float16* h_agent = (_Float16*)ws;                     // 32768*1024*2 = 64MB
  _Float16* Wt_aa = (_Float16*)(ws + (size_t)BATCH * AGENTS * D * 2);
  _Float16* Wt_sp = Wt_aa + (size_t)D * D;               // 2MB each
  float* pooled = (float*)(Wt_sp + (size_t)D * D);       // 1MB
  _Float16* h_scene = (_Float16*)(pooled + (size_t)BATCH * D);  // 0.5MB

  const int M_agent = BATCH * AGENTS;  // 32768

  // 1. weight transpose + f16 convert
  {
    dim3 g(D / 32, D / 32), b(32, 8);
    transpose_cvt_f16<<<g, b, 0, stream>>>(aa_W, Wt_aa);
    transpose_cvt_f16<<<g, b, 0, stream>>>(sp_W, Wt_sp);
  }
  // 2. agent rows: scale*LN*ReLU -> f16
  scale_ln_relu_f16<<<M_agent, 256, 0, stream>>>(encoder_feat, agent_score,
                                                 aa_gamma, aa_beta, h_agent);
  // 3. masked mean pool
  pool_kernel<<<BATCH, 256, 0, stream>>>(encoder_feat, agent_mask, agent_score,
                                         pooled);
  // 4. scene rows: LN*ReLU -> f16
  scale_ln_relu_f16<<<BATCH, 256, 0, stream>>>(pooled, nullptr, sp_gamma,
                                               sp_beta, h_scene);
  // 5. WMMA GEMMs
  gemm_wmma_f16<<<dim3(M_agent / 64, D / 256), 256, 0, stream>>>(
      h_agent, Wt_aa, aa_b, agent_anchor);
  gemm_wmma_f16<<<dim3(BATCH / 64, D / 256), 256, 0, stream>>>(
      h_scene, Wt_sp, sp_b, scene_anchor);
}
